// STPNR_38482906972784
// MI455X (gfx1250) — matrix operand order repaired and checked by reading, not compiled, not run on hardware
//
#include <hip/hip_runtime.h>
#include <hip/hip_bf16.h>

#define BB 32
#define SS 64
#define II 256
#define HH 256
#define OO 128
#define FF 512                 // HH + II
#define TWROW (FF + 8)         // padded row stride in u16 elems (1040 B -> 4-bank shift/row)
#define NTHREADS 512           // 16 wave32 per workgroup

typedef unsigned short u16;
typedef unsigned long long u64;
typedef __attribute__((ext_vector_type(16))) u16    v16u;
typedef __attribute__((ext_vector_type(8)))  u16    v8u;
typedef __attribute__((ext_vector_type(16))) __bf16 v16bf;
typedef __attribute__((ext_vector_type(8)))  float  v8f;
typedef __attribute__((ext_vector_type(4)))  unsigned int tdm_g0_t;
typedef __attribute__((ext_vector_type(8)))  int          tdm_g1_t;
typedef __attribute__((ext_vector_type(4)))  int          tdm_g2_t;
typedef __attribute__((ext_vector_type(8)))  int          tdm_g4_t;

struct Pair16 { v8u lo, hi; };   // 32 bytes == v16bf

#define TW_BYTES (HH * TWROW * 2)                                   // 266240
#define SMEM_BYTES (TW_BYTES + FF*4 + FF*2 + HH*2 + 5*HH*4)         // 274944 < 320K
#define WS_PACKED_BYTES ((size_t)HH * FF * 8)                       // 1 MB packed {lam,gam,W}
#define WS_WBF_BYTES    ((size_t)HH * FF * 2)                       // 256 KB bf16(W) for TDM

__device__ __forceinline__ u16 f2b(float x) {
    union { float f; unsigned u; } v; v.f = x;
    unsigned r = (v.u + 0x7FFFu + ((v.u >> 16) & 1u)) >> 16;        // RNE
    return (u16)r;
}
__device__ __forceinline__ float b2f(u16 b) {
    union { unsigned u; float f; } v; v.u = ((unsigned)b) << 16;
    return v.f;
}
__device__ __forceinline__ v16bf ldfrag(const u16* p0, const u16* p1) {
    Pair16 p; p.lo = *(const v8u*)p0; p.hi = *(const v8u*)p1;
    return __builtin_bit_cast(v16bf, p);
}

// one-time pack: per element {bf16(lam), bf16(gam), fp32 W} -> u64, plus bf16(W) array for TDM
__global__ void stpn_prep(const float* __restrict__ W, const float* __restrict__ lam,
                          const float* __restrict__ gam, u64* __restrict__ packed,
                          u16* __restrict__ wbf) {
    int i = blockIdx.x * blockDim.x + threadIdx.x;
    if (i < HH * FF) {
        union { float f; unsigned u; } wu; wu.f = W[i];
        packed[i] = (u64)f2b(lam[i]) | ((u64)f2b(gam[i]) << 16) | ((u64)wu.u << 32);
        wbf[i] = f2b(W[i]);
    }
}

__global__ __launch_bounds__(NTHREADS, 1)
void stpn_kernel(const float* __restrict__ x,   const float* __restrict__ W,
                 const float* __restrict__ lam, const float* __restrict__ gam,
                 const float* __restrict__ bias,const float* __restrict__ ow,
                 const float* __restrict__ ob,  float* __restrict__ out,
                 const u64* __restrict__ packed, const u16* __restrict__ wbf,
                 int use_packed) {
    extern __shared__ __align__(16) unsigned char smem[];
    u16*   tw   = (u16*)smem;                    // [HH][TWROW] bf16 of (W + f)
    float* u_f  = (float*)(smem + TW_BYTES);     // [FF] total_input fp32
    u16*   u_bf = (u16*)(u_f + FF);              // [FF] total_input bf16
    u16*   hb   = u_bf + FF;                     // [HH] h_fin bf16 (final GEMM)
    float* hcur = (float*)(hb + HH);             // [HH]
    float* hnew = hcur + HH;                     // [HH]
    float* rinv = hnew + HH;                     // [HH] 1/norm
    float* pre  = rinv + HH;                     // [HH]
    float* nsq  = pre  + HH;                     // [HH]

    const int tid  = threadIdx.x;
    const int b    = blockIdx.x;
    const int lane = tid & 31;
    const int wv   = tid >> 5;
    const int r    = lane & 15;
    const int half = lane >> 4;

    // ---- init on-chip state: tw = bf16(W) (f0 = 0) ; h = 0 ----
#if defined(__has_builtin) && __has_builtin(__builtin_amdgcn_tensor_load_to_lds)
    if (use_packed) {
        if (wv == 0) {
            // TDM: 2D tile 512x256 of 2-byte elems, LDS row pad 4 DWORDs every 256 DWORDs
            u64 ga = (u64)(uintptr_t)wbf;
            tdm_g0_t g0;
            g0.x = 1u;                                        // count=1, user descriptor
            g0.y = 0u;                                        // lds_addr = 0 (tw base)
            g0.z = (unsigned)(ga & 0xFFFFFFFFu);              // global_addr lo
            g0.w = (unsigned)((ga >> 32) & 0x01FFFFFFu) | (2u << 30);  // addr hi | type=2
            tdm_g1_t g1;
            g1[0] = (int)((1u << 16) | (1u << 20) | (7u << 22) | (3u << 25));
                    // data_size=2B, pad_enable, pad_interval=256 DW, pad_amount=4 DW
            g1[1] = (int)(0x0200u << 16);                     // tensor_dim0 = 512
            g1[2] = (int)(0x0100u << 16);                     // tensor_dim1 = 256
            g1[3] = (int)(0x0200u << 16);                     // tile_dim0 = 512
            g1[4] = 256;                                      // tile_dim1 = 256, tile_dim2 = 0
            g1[5] = 512;                                      // tensor_dim0_stride = 512
            g1[6] = 0; g1[7] = 0;
            tdm_g2_t z4 = {0, 0, 0, 0};
            tdm_g4_t z8 = {0, 0, 0, 0, 0, 0, 0, 0};
            __builtin_amdgcn_tensor_load_to_lds(g0, g1, z4, z4, z8, 0);
            __builtin_amdgcn_s_wait_tensorcnt(0);
        }
    } else
#endif
    {
        for (int idx = tid; idx < HH * FF; idx += NTHREADS) {
            int h = idx >> 9, f = idx & (FF - 1);
            tw[h * TWROW + f] = f2b(W[idx]);
        }
    }
    if (tid < HH) hcur[tid] = 0.0f;
    __syncthreads();

    const float* xb = x + (size_t)b * SS * II;

    for (int t = 0; t < SS; ++t) {
        // ---- total_input u = [x_t ; h] ----
        float uv = (tid < II) ? xb[t * II + tid] : hcur[tid - II];
        u_f[tid]  = uv;
        u_bf[tid] = f2b(uv);
        __syncthreads();

        // ---- pre = tw·u  and  nsq = diag(tw·twᵀ)  via WMMA, 16 h-rows per wave ----
        const int bm = wv * 16;
        v8f accP = {}; v8f accN = {};
        const u16* rowp = tw + (bm + r) * TWROW;   // A-row == Bᵀ-row (n == r == lane&15)
        #pragma unroll
        for (int kc = 0; kc < FF / 32; ++kc) {
            const int bk = kc * 32;
            v16bf a  = ldfrag(rowp + bk + 16 * half, rowp + bk + 16 * half + 8);
            v16bf bt = ldfrag(rowp + bk + 16 * half, rowp + bk + 16 * half + 8);
            v16bf ub = ldfrag(u_bf + bk + 16 * half, u_bf + bk + 16 * half + 8);
            accP = __builtin_amdgcn_wmma_f32_16x16x32_bf16(false, a, false, ub, (short)0, accP, false, false);
            accN = __builtin_amdgcn_wmma_f32_16x16x32_bf16(false, a, false, bt, (short)0, accN, false, false);
        }
        // C/D layout: lane L holds rows j+8*(L/16), col L%16
        if (lane == 0)  { for (int j = 0; j < 8; ++j) pre[bm + j]     = accP[j]; }
        if (lane == 16) { for (int j = 0; j < 8; ++j) pre[bm + 8 + j] = accP[j]; }
        if (lane < 8)   nsq[bm + lane]            = accN[lane];        // diag rows 0..7
        if (lane >= 24) nsq[bm + 8 + (lane - 24)] = accN[lane - 24];   // diag rows 8..15
        __syncthreads();

        // ---- norm, h_new ----
        if (tid < HH) {
            float nv = sqrtf(nsq[tid]) + 1e-16f;
            hnew[tid] = tanhf(pre[tid] / nv + bias[tid]);
            rinv[tid] = 1.0f / nv;
        }
        __syncthreads();

        // ---- fast-weight update: thread owns column f = tid ----
        {
            float uq = u_f[tid];
            if (use_packed) {
                #pragma unroll 4
                for (int h = 0; h < HH; ++h) {
                    u64 p = packed[h * FF + tid];                     // one b64 load
                    float lamv = b2f((u16)p);
                    float gamv = b2f((u16)(p >> 16));
                    union { unsigned u; float f; } wu; wu.u = (unsigned)(p >> 32);
                    float fo = b2f(tw[h * TWROW + tid]) - wu.f;
                    float fn = lamv * (fo * rinv[h]) + gamv * hnew[h] * uq;
                    tw[h * TWROW + tid] = f2b(wu.f + fn);
                }
            } else {
                #pragma unroll 4
                for (int h = 0; h < HH; ++h) {
                    int   gi  = h * FF + tid;
                    float wvl = W[gi];
                    float fo  = b2f(tw[h * TWROW + tid]) - wvl;
                    float fn  = lam[gi] * (fo * rinv[h]) + gam[gi] * hnew[h] * uq;
                    tw[h * TWROW + tid] = f2b(wvl + fn);
                }
            }
        }
        if (t + 1 < SS && tid < 16)
            __builtin_prefetch(xb + (t + 1) * II + tid * 16, 0, 0);    // global_prefetch_b8
        if (tid < HH) hcur[tid] = hnew[tid];
        __syncthreads();
    }

    // ---- h_fin ----
    if (tid < HH) {
        out[BB * OO + b * HH + tid] = hcur[tid];
        hb[tid] = f2b(hcur[tid]);
    }
    // ---- f_fin = tw - W ----
    for (int idx = tid; idx < HH * FF; idx += NTHREADS) {
        int h = idx >> 9, f = idx & (FF - 1);
        out[BB * OO + BB * HH + (size_t)b * HH * FF + idx] = b2f(tw[h * TWROW + f]) - W[idx];
    }
    __syncthreads();

    // ---- tag_space = h_fin @ out_wᵀ + out_b  via WMMA (waves 0..7, 16 o-rows each) ----
    if (wv < 8) {
        const int bo = wv * 16;
        v8f acc = {};
        #pragma unroll
        for (int kc = 0; kc < HH / 32; ++kc) {
            const int bk = kc * 32;
            v16u aw = {};
            #pragma unroll
            for (int e = 0; e < 16; ++e) {
                int k = bk + 16 * half + e;
                aw[e] = f2b(ow[(bo + r) * HH + k]);
            }
            v16bf a  = __builtin_bit_cast(v16bf, aw);
            v16bf hv = ldfrag(hb + bk + 16 * half, hb + bk + 16 * half + 8);
            acc = __builtin_amdgcn_wmma_f32_16x16x32_bf16(false, a, false, hv, (short)0, acc, false, false);
        }
        if (lane == 0)  for (int j = 0; j < 8; ++j) out[(size_t)b * OO + bo + j]     = acc[j] + ob[bo + j];
        if (lane == 16) for (int j = 0; j < 8; ++j) out[(size_t)b * OO + bo + 8 + j] = acc[j] + ob[bo + 8 + j];
    }
}

extern "C" void kernel_launch(void* const* d_in, const int* in_sizes, int n_in,
                              void* d_out, int out_size, void* d_ws, size_t ws_size,
                              hipStream_t stream) {
    const float* x    = (const float*)d_in[0];
    const float* W    = (const float*)d_in[1];
    const float* lam  = (const float*)d_in[2];
    const float* gam  = (const float*)d_in[3];
    const float* bias = (const float*)d_in[4];
    const float* ow   = (const float*)d_in[5];
    const float* ob   = (const float*)d_in[6];
    float* out = (float*)d_out;

    static_assert(SMEM_BYTES <= 320 * 1024, "LDS budget");

    const int use_packed = (d_ws != nullptr && ws_size >= WS_PACKED_BYTES + WS_WBF_BYTES) ? 1 : 0;
    u64* packed = (u64*)d_ws;
    u16* wbf    = (u16*)((char*)d_ws + WS_PACKED_BYTES);

    if (use_packed) {
        stpn_prep<<<(HH * FF + 255) / 256, 256, 0, stream>>>(W, lam, gam, packed, wbf);
    }
    (void)hipFuncSetAttribute(reinterpret_cast<const void*>(stpn_kernel),
                              hipFuncAttributeMaxDynamicSharedMemorySize, SMEM_BYTES);
    stpn_kernel<<<BB, NTHREADS, SMEM_BYTES, stream>>>(x, W, lam, gam, bias, ow, ob, out,
                                                      packed, wbf, use_packed);
}
